// ModulatedConv2D_33844342293175
// MI455X (gfx1250) — matrix operand order repaired and checked by reading, not compiled, not run on hardware
//
#include <hip/hip_runtime.h>

// ---------------------------------------------------------------------------
// ModulatedConv2D (StyleGAN2): B=8, H=W=64, C=F=512, 3x3 SAME
//   y[b] = demod[b,f] * conv2d(x[b] * (style[b]+1), kernel)
// Implicit GEMM per batch: [4096 x 4608] x [4608 x 512], bf16 WMMA, f32 acc.
// B operand staged in LDS via async global->LDS DMA, double buffered.
// Wave tile 64x64 (acc[4][4]) -> 4x reuse on both A (global) and B (LDS).
// ---------------------------------------------------------------------------

#define Bq   8
#define HH   64
#define WW   64
#define CC   512
#define FF   512
#define KTAP 9
#define KTOT (KTAP * CC)      // 4608
#define HP   66               // padded
#define WP   66

#define STAGE_K   64          // K elements staged per double-buffer stage
#define BROW      72          // padded LDS row stride (elements) -> 144B, conflict-free
#define NSTAGE    (KTAP * (CC / STAGE_K))   // 72

typedef __bf16 bf16x8  __attribute__((ext_vector_type(8)));
typedef __bf16 bf16x16 __attribute__((ext_vector_type(16)));
typedef float  f32x8   __attribute__((ext_vector_type(8)));

union BPack { bf16x16 v; bf16x8 h[2]; };

__device__ __forceinline__ unsigned short f2bf(float f) {
    unsigned int u = __float_as_uint(f);
    unsigned int r = u + 0x7FFFu + ((u >> 16) & 1u);   // RNE
    return (unsigned short)(r >> 16);
}

// A operand (global): lane holds K runs {0..7} and {16..23} (lane>=16: +8):
// two 16B loads at +0 and +16 elements (lane base already includes kg*8).
__device__ __forceinline__ bf16x16 load_a32(const __bf16* p) {
    BPack u;
    u.h[0] = *reinterpret_cast<const bf16x8*>(p);
    u.h[1] = *reinterpret_cast<const bf16x8*>(p + 16);
    return u.v;
}
// B operand (LDS): lane holds K runs {0..7} and {16..23} (base includes kg*8).
__device__ __forceinline__ bf16x16 load_b32_lds(const unsigned short* p) {
    BPack u;
    u.h[0] = *reinterpret_cast<const bf16x8*>(p);
    u.h[1] = *reinterpret_cast<const bf16x8*>(p + 16);
    return u.v;
}

__device__ __forceinline__ f32x8 wmma_bf16(bf16x16 a, bf16x16 b, f32x8 c) {
    return __builtin_amdgcn_wmma_f32_16x16x32_bf16(
        false, a, false, b, (short)0, c, false, false);
}

// ---- 1) scale input by (style+1), convert to bf16, zero-pad to [B,66,66,C]
__global__ void scale_pad_kernel(const float* __restrict__ x,
                                 const float* __restrict__ style,
                                 unsigned short* __restrict__ xpad, int total) {
    int i = blockIdx.x * blockDim.x + threadIdx.x;
    if (i >= total) return;
    int c  = i & (CC - 1);
    int t  = i >> 9;
    int wp = t % WP;
    int t2 = t / WP;
    int hp = t2 % HP;
    int b  = t2 / HP;
    unsigned short v = 0;
    if (hp >= 1 && hp <= HH && wp >= 1 && wp <= WW) {
        float s   = style[b * CC + c] + 1.0f;
        float val = x[(((b * HH + (hp - 1)) * WW + (wp - 1)) << 9) + c] * s;
        v = f2bf(val);
    }
    xpad[i] = v;
}

// ---- 2) weights [3,3,C,F] f32  ->  wt[F][tap*C + c] bf16 (transposed)
__global__ void pack_w_kernel(const float* __restrict__ kern,
                              unsigned short* __restrict__ wt, int total) {
    int i = blockIdx.x * blockDim.x + threadIdx.x;
    if (i >= total) return;
    int k = i % KTOT;                 // tap*512 + c
    wt[i] = f2bf(kern[k * FF + (i / KTOT)]);
}

// ---- 3) K2[c,f] = sum_tap kernel[tap,c,f]^2
__global__ void k2_kernel(const float* __restrict__ kern, float* __restrict__ K2) {
    int i = blockIdx.x * blockDim.x + threadIdx.x;   // i = c*512 + f
    float s = 0.f;
#pragma unroll
    for (int tap = 0; tap < KTAP; ++tap) {
        float v = kern[tap * (CC * FF) + i];
        s += v * v;
    }
    K2[i] = s;
}

// ---- 4) demod[b,f] = rsqrt(sum_c (style+1)^2 * K2[c,f] + eps)
__global__ void demod_kernel(const float* __restrict__ style,
                             const float* __restrict__ K2,
                             float* __restrict__ dmd) {
    int i = blockIdx.x * blockDim.x + threadIdx.x;   // b*512 + f
    int f = i & (FF - 1);
    int b = i >> 9;
    float sum = 0.f;
    for (int c = 0; c < CC; ++c) {
        float s = style[b * CC + c] + 1.0f;
        sum += s * s * K2[c * FF + f];
    }
    dmd[i] = rsqrtf(sum + 1e-8f);
}

// Issue async global->LDS copies for one B stage:
// LDS tile [128 filters][STAGE_K] (row stride BROW elems), 16KB = 256thr x 4 x b128.
__device__ __forceinline__ void fill_B(unsigned int ldsOffBase, const __bf16* wblk,
                                       int tid, int stageOff /*elements*/) {
    const int kp = tid & 7;           // 8 x 8-element chunks per row
    const int fb = tid >> 3;          // 0..31
#pragma unroll
    for (int t = 0; t < 4; ++t) {
        int f = t * 32 + fb;          // 0..127
        unsigned int loff = ldsOffBase + (unsigned int)(f * BROW + kp * 8) * 2u;
        unsigned int goff = (unsigned int)(f * KTOT + stageOff + kp * 8) * 2u;
        asm volatile("global_load_async_to_lds_b128 %0, %1, %2"
                     :: "v"(loff), "v"(goff), "s"(wblk) : "memory");
    }
}

// ---- 5) main conv as implicit GEMM with bf16 WMMA + async LDS staging of B
// grid (16, 4, 8): x = pixel block (256 px), y = filter block (128 f), z = batch
// 256 threads = 8 waves; wave grid 4(M) x 2(N); wave tile 64 x 64.
__global__ __launch_bounds__(256) void conv_wmma_kernel(
    const __bf16* __restrict__ xpad,   // [B][66][66][512]
    const __bf16* __restrict__ wt,     // [512][4608]
    const float*  __restrict__ dmd,    // [B][512]
    float* __restrict__ out)           // [B][4096][512]
{
    __shared__ __align__(16) unsigned short ldsB[2][128 * BROW];

    const int tid  = threadIdx.x;
    const int lane = tid & 31;
    const int wid  = tid >> 5;
    const int wm   = wid & 3;          // 0..3 along M (64 px each)
    const int wn   = wid >> 2;         // 0..1 along N (64 f each)
    const int b    = blockIdx.z;
    const int lr   = lane & 15;
    const int kg   = lane >> 4;        // half-wave K group

    const int pix0 = blockIdx.x * 256 + wm * 64;
    const int f0   = blockIdx.y * 128 + wn * 64;

    const __bf16* xb = xpad + (size_t)b * (HP * WP * CC);

    // Per-lane A row bases (pixel rows), includes kg*8 channel offset
    const __bf16* aRow[4];
#pragma unroll
    for (int mt = 0; mt < 4; ++mt) {
        int p = pix0 + mt * 16 + lr;
        int h = p >> 6, w = p & 63;
        aRow[mt] = xb + ((h * WP + w) << 9) + kg * 8;
    }

    // Workgroup's weight block base (uniform -> SGPR for async saddr form)
    const __bf16* wblk = wt + (size_t)(blockIdx.y * 128) * KTOT;
    const unsigned int ldsBase = (unsigned int)(uintptr_t)(&ldsB[0][0]);
    const unsigned int bufStride = (unsigned int)(128 * BROW * 2);

    f32x8 acc[4][4] = {};

    // Prologue: stage 0 into buffer 0
    fill_B(ldsBase, wblk, tid, 0);
    asm volatile("s_wait_asynccnt 0" ::: "memory");
    __syncthreads();

#pragma unroll 1
    for (int st = 0; st < NSTAGE; ++st) {
        const int buf = st & 1;
        if (st + 1 < NSTAGE) {
            const int nst  = st + 1;
            const int ntap = nst >> 3;
            const int nc0  = (nst & 7) << 6;
            fill_B(ldsBase + (unsigned int)(nst & 1) * bufStride, wblk, tid,
                   ntap * CC + nc0);
        }
        const int tap = st >> 3;
        const int c0  = (st & 7) << 6;
        const int tapA = (((tap / 3) * WP + (tap % 3)) << 9);
        const __bf16* aP0 = aRow[0] + tapA + c0;
        const __bf16* aP1 = aRow[1] + tapA + c0;
        const __bf16* aP2 = aRow[2] + tapA + c0;
        const __bf16* aP3 = aRow[3] + tapA + c0;
        const unsigned short* Bbase =
            &ldsB[buf][(wn * 64 + lr) * BROW + kg * 8];

#pragma unroll
        for (int s = 0; s < 2; ++s) {           // two 32-K WMMA steps per stage
            bf16x16 Av[4];
            Av[0] = load_a32(aP0 + s * 32);
            Av[1] = load_a32(aP1 + s * 32);
            Av[2] = load_a32(aP2 + s * 32);
            Av[3] = load_a32(aP3 + s * 32);
            bf16x16 Bv[4];
#pragma unroll
            for (int nt = 0; nt < 4; ++nt)
                Bv[nt] = load_b32_lds(Bbase + nt * 16 * BROW + s * 32);
#pragma unroll
            for (int mt = 0; mt < 4; ++mt)
#pragma unroll
                for (int nt = 0; nt < 4; ++nt)
                    acc[mt][nt] = wmma_bf16(Av[mt], Bv[nt], acc[mt][nt]);
        }

        // Next buffer fully written + all waves done reading current buffer.
        asm volatile("s_wait_asynccnt 0" ::: "memory");
        __syncthreads();
    }

    // Epilogue: apply per-(b,f) demodulation, store f32.
    // C/D layout: VGPR r, lanes 0-15 -> M=r, N=lane; lanes 16-31 -> M=r+8.
#pragma unroll
    for (int nt = 0; nt < 4; ++nt) {
        int f = f0 + nt * 16 + lr;
        float dm = dmd[b * FF + f];
#pragma unroll
        for (int mt = 0; mt < 4; ++mt) {
            int pbase = pix0 + mt * 16 + kg * 8;
            float* o = out + (((size_t)b * (HH * WW) + pbase) << 9) + f;
#pragma unroll
            for (int r = 0; r < 8; ++r)
                o[(size_t)r * FF] = acc[mt][nt][r] * dm;
        }
    }
}

// ---------------------------------------------------------------------------
extern "C" void kernel_launch(void* const* d_in, const int* in_sizes, int n_in,
                              void* d_out, int out_size, void* d_ws, size_t ws_size,
                              hipStream_t stream) {
    const float* x     = (const float*)d_in[0];   // [8,64,64,512]
    const float* style = (const float*)d_in[1];   // [8,512]
    const float* kern  = (const float*)d_in[2];   // [3,3,512,512]
    float* out = (float*)d_out;

    // workspace layout (16B aligned)
    const size_t XPAD_ELEMS = (size_t)Bq * HP * WP * CC;     // 17,842,176
    const size_t WT_ELEMS   = (size_t)FF * KTOT;             //  2,359,296
    char* ws = (char*)d_ws;
    unsigned short* xpad_u = (unsigned short*)ws;
    unsigned short* wt_u   = (unsigned short*)(ws + XPAD_ELEMS * 2);
    float* K2  = (float*)(ws + XPAD_ELEMS * 2 + WT_ELEMS * 2);
    float* dmd = (float*)(ws + XPAD_ELEMS * 2 + WT_ELEMS * 2 + (size_t)CC * FF * 4);

    {   // 1) scale + pad + bf16 convert
        int total = (int)XPAD_ELEMS;
        scale_pad_kernel<<<(total + 255) / 256, 256, 0, stream>>>(x, style, xpad_u, total);
    }
    {   // 2) transpose weights to [F][K], bf16
        int total = (int)WT_ELEMS;
        pack_w_kernel<<<(total + 255) / 256, 256, 0, stream>>>(kern, wt_u, total);
    }
    // 3) K2 = sum of squared taps
    k2_kernel<<<(CC * FF) / 256, 256, 0, stream>>>(kern, K2);
    // 4) demod factors
    demod_kernel<<<(Bq * FF) / 256, 256, 0, stream>>>(style, K2, dmd);
    // 5) WMMA conv with async-LDS staged weights
    dim3 grid((HH * WW) / 256, FF / 128, Bq);
    conv_wmma_kernel<<<grid, 256, 0, stream>>>(
        (const __bf16*)xpad_u, (const __bf16*)wt_u, dmd, out);
}
